// Retention_8985071583703
// MI455X (gfx1250) — matrix operand order repaired and checked by reading, not compiled, not run on hardware
//
#include <hip/hip_runtime.h>
#include <hip/hip_bf16.h>

// ---------- problem constants ----------
#define BB   8
#define TT   768
#define DD   1024
#define HH   16
#define DH   64
#define ROWS (BB * TT)     // 6144
#define ND3  (3 * DD)      // 3072

typedef __attribute__((ext_vector_type(16))) _Float16 v16h;
typedef __attribute__((ext_vector_type(8)))  _Float16 v8h;
typedef __attribute__((ext_vector_type(8)))  float    v8f;

// D = A(16x32) * B(32x16) + C, f16 inputs, f32 accumulate
__device__ __forceinline__ v8f wmma_f16(v16h a, v16h b, v8f c) {
  return __builtin_amdgcn_wmma_f32_16x16x32_f16(
      /*neg_a=*/false, a, /*neg_b=*/false, b,
      /*c_mod=*/(short)0, c, /*reuse_a=*/false, /*reuse_b=*/false);
}

// A-fragment (16x32, 16-bit): lane half h, m = lane&15.
// row = row0+m ; chunk0 = cols [h*8 .. h*8+7], chunk1 = cols [16+h*8 ..]
__device__ __forceinline__ v16h load_afrag(const _Float16* __restrict__ src,
                                           int ld, int row0, int col0, int lane) {
  int half = lane >> 4, m = lane & 15;
  const _Float16* p = src + (size_t)(row0 + m) * ld + col0 + half * 8;
  union { v16h v; v8h h2[2]; } u;
  u.h2[0] = *(const v8h*)(p);
  u.h2[1] = *(const v8h*)(p + 16);
  return u.v;
}

// B-fragment (32x16, 16-bit) where B[k][n] = src[n][k] (src row-major over n):
// lane: n = n0 + (lane&15), k = k0 + half*8 .. +7 and k0 + 16 + half*8 .. +7.
__device__ __forceinline__ v16h load_bfrag(const _Float16* __restrict__ src,
                                           int ld, int n0, int k0, int lane) {
  int half = lane >> 4, n = lane & 15;
  const _Float16* p = src + (size_t)(n0 + n) * ld + k0 + half * 8;
  union { v16h v; v8h h2[2]; } u;
  u.h2[0] = *(const v8h*)(p);
  u.h2[1] = *(const v8h*)(p + 16);
  return u.v;
}

// ---------- kernel 0: fp32 -> f16 convert of x and W ----------
__global__ void cvt_f32_to_f16(const float* __restrict__ x,
                               const float* __restrict__ w,
                               _Float16* __restrict__ xb,
                               _Float16* __restrict__ wb,
                               int nx, int nw) {
  int i = blockIdx.x * blockDim.x + threadIdx.x;
  if (i < nx)            xb[i]      = (_Float16)x[i];
  else if (i < nx + nw)  wb[i - nx] = (_Float16)w[i - nx];
}

// ---------- kernel 1: qkv GEMM + bias + RoPE + head-major scatter ----------
// Each wave: 32(M) x 64(N) tile; B-fragments reused across both 16-row
// M-subtiles (1.5 loads/WMMA). N strip == one head of one of {q,k,v}.
__global__ void qkv_rope_kernel(const float* __restrict__ bias,
                                const _Float16* __restrict__ xb,   // (6144,1024)
                                const _Float16* __restrict__ Wb,   // (3072,1024)
                                _Float16* __restrict__ qh,         // (B,H,T,64)
                                _Float16* __restrict__ kh,         // (B,H,T,64)
                                _Float16* __restrict__ vt) {       // (B,H,64,T)
  const int lane = threadIdx.x & 31;
  const int wave = threadIdx.x >> 5;
  const int wid  = blockIdx.x * 4 + wave;
  const int mtile  = wid % (ROWS / 32);      // 0..191 (block shares nstrip -> W hits WGP$)
  const int nstrip = wid / (ROWS / 32);      // 0..47
  const int row0 = mtile * 32;
  const int j0   = nstrip * 64;

  v8f c[2][4];
#pragma unroll
  for (int mi = 0; mi < 2; ++mi)
#pragma unroll
    for (int ci = 0; ci < 4; ++ci) c[mi][ci] = {};

  for (int k = 0; k < DD; k += 32) {
    const v16h a0 = load_afrag(xb, DD, row0,      k, lane);
    const v16h a1 = load_afrag(xb, DD, row0 + 16, k, lane);
#pragma unroll
    for (int ci = 0; ci < 4; ++ci) {
      const v16h b = load_bfrag(Wb, DD, j0 + 16 * ci, k, lane);
      c[0][ci] = wmma_f16(a0, b, c[0][ci]);
      c[1][ci] = wmma_f16(a1, b, c[1][ci]);
    }
  }

  const int n = lane & 15, half = lane >> 4;
  const float b0 = bias[j0 + n],      b1 = bias[j0 + 16 + n];
  const float b2 = bias[j0 + 32 + n], b3 = bias[j0 + 48 + n];
#pragma unroll
  for (int mi = 0; mi < 2; ++mi)
#pragma unroll
    for (int r = 0; r < 8; ++r) {
      c[mi][0][r] += b0; c[mi][1][r] += b1;
      c[mi][2][r] += b2; c[mi][3][r] += b3;
    }

  const int seg  = j0 / DD;              // 0=q 1=k 2=v
  const int head = (j0 % DD) / DH;
  const int bidx = row0 / TT;            // tile never crosses batch (768 % 32 == 0)

  if (seg < 2) {
    // RoPE: pair (dh, dh+32) lives in (c0,c2)/(c1,c3) of the SAME lane.
    // inv_freq[fi] = 10000^(-fi/32) = exp(-fi * ln(1e4)/32)
    const float kf  = 0.28782313662425574f;         // ln(10000)/32
    const float if0 = __expf(-kf * (float)n);
    const float if1 = __expf(-kf * (float)(n + 16));
    _Float16* dst = (seg == 0 ? qh : kh) + (size_t)(bidx * HH + head) * TT * DH;
#pragma unroll
    for (int mi = 0; mi < 2; ++mi) {
      const int tbase = row0 % TT + 16 * mi;
#pragma unroll
      for (int r = 0; r < 8; ++r) {
        const int   t  = tbase + r + half * 8;
        const float tf = (float)t;
        const float a0 = tf * if0, a1 = tf * if1;
        const float cs0 = __cosf(a0), sn0 = __sinf(a0);
        const float cs1 = __cosf(a1), sn1 = __sinf(a1);
        const float q0 = c[mi][0][r], q1 = c[mi][1][r];
        const float q2 = c[mi][2][r], q3 = c[mi][3][r];
        _Float16* p = dst + (size_t)t * DH;
        p[n]      = (_Float16)(q0 * cs0 - q2 * sn0);   // dh = n
        p[32 + n] = (_Float16)(q2 * cs0 + q0 * sn0);   // dh = n + 32
        p[16 + n] = (_Float16)(q1 * cs1 - q3 * sn1);   // dh = n + 16
        p[48 + n] = (_Float16)(q3 * cs1 + q1 * sn1);   // dh = n + 48
      }
    }
  } else {
    // V stored transposed (dh, t) so attention B-frags are contiguous.
    _Float16* dst = vt + (size_t)(bidx * HH + head) * DH * TT;
#pragma unroll
    for (int mi = 0; mi < 2; ++mi) {
      const int tbase = row0 % TT + 16 * mi;
#pragma unroll
      for (int r = 0; r < 8; ++r) {
        const int t = tbase + r + half * 8;
        dst[(size_t)(n)      * TT + t] = (_Float16)c[mi][0][r];
        dst[(size_t)(16 + n) * TT + t] = (_Float16)c[mi][1][r];
        dst[(size_t)(32 + n) * TT + t] = (_Float16)c[mi][2][r];
        dst[(size_t)(48 + n) * TT + t] = (_Float16)c[mi][3][r];
      }
    }
  }
}

// ---------- kernel 2: causal retention  O = tril(Q K^T) V ----------
// One wave per (b,h, 32-row t-tile); K and V fragments reused across both
// 16-row M-subtiles (1.0 loads/WMMA). s loops only over s <= t (causal).
__global__ void retention_attn_kernel(const _Float16* __restrict__ qh,
                                      const _Float16* __restrict__ kh,
                                      const _Float16* __restrict__ vt,
                                      float* __restrict__ out) {
  __shared__ _Float16 pstage[4][2][16 * 32] __attribute__((aligned(32)));
  const int lane = threadIdx.x & 31;
  const int wave = threadIdx.x >> 5;
  const int wid  = blockIdx.x * 4 + wave;
  const int tt   = wid % (TT / 32);     // 0..23 (block shares (b,h) -> K/V hit WGP$)
  const int bh   = wid / (TT / 32);     // 0..127
  const int b    = bh >> 4, h = bh & 15;
  const int t0   = tt * 32;

  const _Float16* Q = qh + (size_t)bh * TT * DH;
  const _Float16* K = kh + (size_t)bh * TT * DH;
  const _Float16* V = vt + (size_t)bh * DH * TT;   // (dh, t)

  v16h aq[2][2];
#pragma unroll
  for (int mi = 0; mi < 2; ++mi) {
    aq[mi][0] = load_afrag(Q, DH, t0 + 16 * mi, 0,  lane);
    aq[mi][1] = load_afrag(Q, DH, t0 + 16 * mi, 32, lane);
  }

  v8f o[2][4];
#pragma unroll
  for (int mi = 0; mi < 2; ++mi)
#pragma unroll
    for (int ci = 0; ci < 4; ++ci) o[mi][ci] = {};

  const int n = lane & 15, half = lane >> 4;

  for (int s0 = 0; s0 < t0 + 32; s0 += 32) {
    // ---- scores: 2 M-subtiles x 2 s-subtiles, K-dim = DH(64) in two steps ----
    v8f cs[2][2];
#pragma unroll
    for (int sj = 0; sj < 2; ++sj) {
      const v16h bk0 = load_bfrag(K, DH, s0 + 16 * sj, 0,  lane);
      const v16h bk1 = load_bfrag(K, DH, s0 + 16 * sj, 32, lane);
#pragma unroll
      for (int mi = 0; mi < 2; ++mi) {
        v8f acc = {};
        acc = wmma_f16(aq[mi][0], bk0, acc);
        acc = wmma_f16(aq[mi][1], bk1, acc);
        cs[mi][sj] = acc;
      }
    }

    // ---- causal mask on diagonal blocks (per element: keep s <= t) ----
#pragma unroll
    for (int mi = 0; mi < 2; ++mi) {
      if (s0 + 31 > t0 + 16 * mi) {
#pragma unroll
        for (int r = 0; r < 8; ++r) {
          const int t = t0 + 16 * mi + r + half * 8;
          if (s0 + n > t)      cs[mi][0][r] = 0.f;
          if (s0 + 16 + n > t) cs[mi][1][r] = 0.f;
        }
      }
    }

    // ---- C-layout -> A-fragment via per-wave LDS tiles (in-order DS ops) ----
    v16h pa[2];
#pragma unroll
    for (int mi = 0; mi < 2; ++mi) {
      _Float16* stg = pstage[wave][mi];
#pragma unroll
      for (int r = 0; r < 8; ++r) {
        const int m = r + half * 8;
        stg[m * 32 + n]      = (_Float16)cs[mi][0][r];
        stg[m * 32 + 16 + n] = (_Float16)cs[mi][1][r];
      }
      const _Float16* p = stg + n * 32 + half * 8;
      union { v16h v; v8h h2[2]; } u;
      u.h2[0] = *(const v8h*)(p);
      u.h2[1] = *(const v8h*)(p + 16);
      pa[mi] = u.v;
    }

    // ---- O += P(16x32) * V(32x64) : 4 N-subtiles, V-frag shared by both P ----
#pragma unroll
    for (int ci = 0; ci < 4; ++ci) {
      const v16h bv = load_bfrag(V, TT, 16 * ci, s0, lane);
      o[0][ci] = wmma_f16(pa[0], bv, o[0][ci]);
      o[1][ci] = wmma_f16(pa[1], bv, o[1][ci]);
    }
  }

  // ---- write fp32 output, layout (B, T, H*64) ----
  float* obase = out + (size_t)b * TT * DD + h * DH;
#pragma unroll
  for (int mi = 0; mi < 2; ++mi) {
#pragma unroll
    for (int r = 0; r < 8; ++r) {
      const int t = t0 + 16 * mi + r + half * 8;
      float* p = obase + (size_t)t * DD;
      p[n]      = o[mi][0][r];
      p[16 + n] = o[mi][1][r];
      p[32 + n] = o[mi][2][r];
      p[48 + n] = o[mi][3][r];
    }
  }
}

// ---------- host launch ----------
extern "C" void kernel_launch(void* const* d_in, const int* in_sizes, int n_in,
                              void* d_out, int out_size, void* d_ws, size_t ws_size,
                              hipStream_t stream) {
  const float* x    = (const float*)d_in[0];   // (B,T,D)
  const float* W    = (const float*)d_in[1];   // (3D,D)
  const float* bias = (const float*)d_in[2];   // (3D,)
  float* out = (float*)d_out;                  // (B,T,D)

  const int nx = ROWS * DD;        // 6,291,456
  const int nw = ND3 * DD;         // 3,145,728

  char* ws = (char*)d_ws;
  _Float16* xb = (_Float16*)(ws);                              // 12,582,912 B
  _Float16* Wb = (_Float16*)(ws + (size_t)12582912);           //  6,291,456 B
  _Float16* qh = (_Float16*)(ws + (size_t)18874368);           // 12,582,912 B
  _Float16* kh = (_Float16*)(ws + (size_t)31457280);           // 12,582,912 B
  _Float16* vt = (_Float16*)(ws + (size_t)44040192);           // 12,582,912 B
  // total workspace use: 56,623,104 bytes

  cvt_f32_to_f16<<<(nx + nw + 255) / 256, 256, 0, stream>>>(x, W, xb, Wb, nx, nw);

  // 192 M-tiles * 48 N-strips = 9216 waves, 4 waves (128 thr) per block
  qkv_rope_kernel<<<(ROWS / 32) * (ND3 / 64) / 4, 128, 0, stream>>>(
      bias, xb, Wb, qh, kh, vt);

  // 128 (b,h) * 24 t-tiles = 3072 waves, 4 waves per block
  retention_attn_kernel<<<(BB * HH) * (TT / 32) / 4, 128, 0, stream>>>(
      qh, kh, vt, out);
}
